// k_hop_GraphNN_74560632258903
// MI455X (gfx1250) — compile-verified, matrix operand order checked by reading, not run on hardware
//
#include <hip/hip_runtime.h>

#define N_NODES  100000
#define N_EDGES  800000
#define N_GRAPHS 256
#define D        128
#define EPS      1e-5f

typedef __attribute__((ext_vector_type(16))) _Float16 v16h;
typedef __attribute__((ext_vector_type(8)))  float    v8f;

__device__ __forceinline__ void atomAddF(float* p, float v) {
  // lowers to global_atomic_add_f32 (device scope, non-returning)
  __hip_atomic_fetch_add(p, v, __ATOMIC_RELAXED, __HIP_MEMORY_SCOPE_AGENT);
}

// K index for half j (0..15) of a 16-bit A/B WMMA fragment (ISA §7.12.2):
// VGPR vg<4 -> K=2vg(+1); vg>=4 -> K=2vg+8(+1); +8 for lanes 16..31.
__device__ __forceinline__ int kmap(int j, int hi8) {
  const int vg = j >> 1;
  return ((vg < 4) ? 2 * vg : 2 * vg + 8) + hi8 + (j & 1);
}

__global__ void k_zero(float* __restrict__ p, int n) {
  int i = blockIdx.x * blockDim.x + threadIdx.x;
  if (i < n) p[i] = 0.f;
}

// One wave per edge: lane handles 4 consecutive features (32 lanes * 4 = 128).
// m[dst] += h[src]
__global__ void k_scatter(const float* __restrict__ h, const int* __restrict__ src,
                          const int* __restrict__ dst, float* __restrict__ m) {
  int e = blockIdx.x * 8 + (threadIdx.x >> 5);
  if (e >= N_EDGES) return;
  int lane = threadIdx.x & 31;
  const float4 v = ((const float4*)(h + (long)src[e] * D))[lane];
  float* p = m + (long)dst[e] * D + lane * 4;
  atomAddF(p + 0, v.x);
  atomAddF(p + 1, v.y);
  atomAddF(p + 2, v.z);
  atomAddF(p + 3, v.w);
}

// Pre-pack W[128,128] (f32, row-major, W[k][n]) into f16 WMMA B-fragments:
// Wp[(kt*8 + colTile)*32 + lane] = lane's v16h B fragment for that K-chunk/tile.
// 4*8*32 = 1024 fragments = 32 KB; launch <<<4,256>>>.
__global__ void k_pack_w(const float* __restrict__ W, v16h* __restrict__ Wp) {
  const int id   = blockIdx.x * blockDim.x + threadIdx.x;  // 0..1023
  const int lane = id & 31;
  const int tile = (id >> 5) & 7;
  const int kt   = id >> 8;
  const int hi8  = (lane & 16) ? 8 : 0;
  const int lr   = lane & 15;
  const int kb   = kt * 32;
  v16h b;
  #pragma unroll
  for (int j = 0; j < 16; ++j) {
    const int k = kb + kmap(j, hi8);
    b[j] = (_Float16)W[(long)k * D + tile * 16 + lr];
  }
  Wp[id] = b;
}

// out[M,128] = relu(A[M,128] @ W + bias) with W pre-packed as B-fragments.
// Block = 256 threads = 8 waves; block covers 32 rows (two 16-row tiles),
// wave w covers cols [16w,16w+16). K=128 -> 4 chunks, 8 WMMAs/wave,
// two independent accumulator chains. M must be a multiple of 32.
__global__ void k_gemm_wmma(const float* __restrict__ A, const v16h* __restrict__ Wp,
                            const float* __restrict__ bias, float* __restrict__ out) {
  const int row0 = blockIdx.x * 32;
  const int wave = threadIdx.x >> 5;       // 0..7 -> column tile
  const int lane = threadIdx.x & 31;
  const int hi8  = (lane & 16) ? 8 : 0;
  const int lr   = lane & 15;
  const int col0 = wave * 16;

  const float* __restrict__ a0row = A + (long)(row0 + lr) * D;
  const float* __restrict__ a1row = A + (long)(row0 + 16 + lr) * D;

  v8f c0 = {}, c1 = {};
  #pragma unroll
  for (int kt = 0; kt < 4; ++kt) {
    const int kb = kt * 32;
    const v16h b = Wp[(kt * 8 + wave) * 32 + lane];   // one 32B contiguous load
    v16h a0, a1;
    #pragma unroll
    for (int j = 0; j < 16; ++j) {
      const int k = kb + kmap(j, hi8);
      a0[j] = (_Float16)a0row[k];
      a1[j] = (_Float16)a1row[k];
    }
    c0 = __builtin_amdgcn_wmma_f32_16x16x32_f16(false, a0, false, b, (short)0, c0, false, false);
    c1 = __builtin_amdgcn_wmma_f32_16x16x32_f16(false, a1, false, b, (short)0, c1, false, false);
  }

  const float bcol = bias[col0 + lr];
  const int   ho   = (lane & 16) ? 8 : 0;   // C/D layout: VGPR r -> M=r (+8 for hi lanes)
  #pragma unroll
  for (int r = 0; r < 8; ++r) {
    out[(long)(row0 + r + ho) * D + col0 + lr]      = fmaxf(c0[r] + bcol, 0.f);
    out[(long)(row0 + 16 + r + ho) * D + col0 + lr] = fmaxf(c1[r] + bcol, 0.f);
  }
}

// Per-column mean + rsqrt(var+eps) over `rows` rows; one block per feature.
__global__ void k_colstats(const float* __restrict__ x, int rows,
                           float* __restrict__ mu, float* __restrict__ rs) {
  const int f = blockIdx.x;
  float s = 0.f, s2 = 0.f;
  for (int r = threadIdx.x; r < rows; r += blockDim.x) {
    float v = x[(long)r * D + f];
    s += v; s2 += v * v;
  }
  __shared__ float sh[256], sh2[256];
  sh[threadIdx.x] = s; sh2[threadIdx.x] = s2;
  __syncthreads();
  for (int st = blockDim.x >> 1; st > 0; st >>= 1) {
    if (threadIdx.x < st) {
      sh[threadIdx.x]  += sh[threadIdx.x + st];
      sh2[threadIdx.x] += sh2[threadIdx.x + st];
    }
    __syncthreads();
  }
  if (threadIdx.x == 0) {
    float m   = sh[0] / (float)rows;
    float var = sh2[0] / (float)rows - m * m;
    mu[f] = m;
    rs[f] = rsqrtf(var + EPS);
  }
}

// pooled[idx[n]] += bn1(h[n])   (bn1 applied on the fly; one wave per node)
__global__ void k_pool(const float* __restrict__ h, const int* __restrict__ idx,
                       const float* __restrict__ mu, const float* __restrict__ rs,
                       const float* __restrict__ gamma, const float* __restrict__ beta,
                       float* __restrict__ pooled) {
  int n = blockIdx.x * 8 + (threadIdx.x >> 5);
  if (n >= N_NODES) return;
  int lane = threadIdx.x & 31;
  const float4 v  = ((const float4*)(h + (long)n * D))[lane];
  const float4 m4 = ((const float4*)mu)[lane];
  const float4 r4 = ((const float4*)rs)[lane];
  const float4 g4 = ((const float4*)gamma)[lane];
  const float4 b4 = ((const float4*)beta)[lane];
  float* p = pooled + (long)idx[n] * D + lane * 4;
  atomAddF(p + 0, (v.x - m4.x) * r4.x * g4.x + b4.x);
  atomAddF(p + 1, (v.y - m4.y) * r4.y * g4.y + b4.y);
  atomAddF(p + 2, (v.z - m4.z) * r4.z * g4.z + b4.z);
  atomAddF(p + 3, (v.w - m4.w) * r4.w * g4.w + b4.w);
}

// bn2 over the tiny [256,128] pooled matrix: one thread per feature does both passes.
__global__ void k_bn_rows(const float* __restrict__ x, const float* __restrict__ gamma,
                          const float* __restrict__ beta, float* __restrict__ y) {
  int f = threadIdx.x;
  float s = 0.f, s2 = 0.f;
  for (int g = 0; g < N_GRAPHS; ++g) {
    float v = x[(long)g * D + f];
    s += v; s2 += v * v;
  }
  float m   = s / (float)N_GRAPHS;
  float var = s2 / (float)N_GRAPHS - m * m;
  float rs  = rsqrtf(var + EPS);
  float ga = gamma[f], be = beta[f];
  for (int g = 0; g < N_GRAPHS; ++g)
    y[(long)g * D + f] = (x[(long)g * D + f] - m) * rs * ga + be;
}

extern "C" void kernel_launch(void* const* d_in, const int* in_sizes, int n_in,
                              void* d_out, int out_size, void* d_ws, size_t ws_size,
                              hipStream_t stream) {
  (void)in_sizes; (void)n_in; (void)out_size; (void)ws_size;
  const float* X    = (const float*)d_in[0];   // final_features [N, 128]
  const int*   esrc = (const int*)  d_in[1];
  const int*   edst = (const int*)  d_in[2];
  const int*   idx  = (const int*)  d_in[3];
  /* d_in[4] (segment) unused by reference */
  const float* W0   = (const float*)d_in[5];
  const float* b0   = (const float*)d_in[6];
  const float* W1   = (const float*)d_in[7];
  const float* b1   = (const float*)d_in[8];
  const float* g1   = (const float*)d_in[9];
  const float* be1  = (const float*)d_in[10];
  const float* g2   = (const float*)d_in[11];
  const float* be2  = (const float*)d_in[12];
  const float* Wf   = (const float*)d_in[13];
  const float* bf   = (const float*)d_in[14];

  char*  ws = (char*)d_ws;
  const size_t ND = (size_t)N_NODES * D;
  float* m       = (float*)ws;                        // [N, D] messages
  float* h       = (float*)(ws + ND * 4);             // [N, D] hidden
  float* aux     = (float*)(ws + ND * 8);
  float* mu1     = aux;                                // [D]
  float* rs1     = aux + D;                            // [D]
  float* pooled  = aux + 2 * D;                        // [G, D]
  float* pooledn = pooled + (size_t)N_GRAPHS * D;      // [G, D]
  v16h*  Wp      = (v16h*)(pooledn + (size_t)N_GRAPHS * D);  // 1024 fragments, 32 KB

  // ---- hop 0: m = A@X ; h = relu(m @ W0 + b0) ----
  k_zero     <<<((int)ND + 255) / 256, 256, 0, stream>>>(m, (int)ND);
  k_scatter  <<<N_EDGES / 8,           256, 0, stream>>>(X, esrc, edst, m);
  k_pack_w   <<<4,                     256, 0, stream>>>(W0, Wp);
  k_gemm_wmma<<<N_NODES / 32,          256, 0, stream>>>(m, Wp, b0, h);

  // ---- hop 1: m = A@h ; h = relu(m @ W1 + b1) ----
  k_zero     <<<((int)ND + 255) / 256, 256, 0, stream>>>(m, (int)ND);
  k_scatter  <<<N_EDGES / 8,           256, 0, stream>>>(h, esrc, edst, m);
  k_pack_w   <<<4,                     256, 0, stream>>>(W1, Wp);
  k_gemm_wmma<<<N_NODES / 32,          256, 0, stream>>>(m, Wp, b1, h);

  // ---- bn1 stats, then fused bn1-apply + graph readout scatter-add ----
  k_colstats<<<D, 256, 0, stream>>>(h, N_NODES, mu1, rs1);
  k_zero    <<<(N_GRAPHS * D + 255) / 256, 256, 0, stream>>>(pooled, N_GRAPHS * D);
  k_pool    <<<(N_NODES + 7) / 8,          256, 0, stream>>>(h, idx, mu1, rs1, g1, be1, pooled);

  // ---- bn2 -> fc1 -> relu ----
  k_bn_rows  <<<1, D, 0, stream>>>(pooled, g2, be2, pooledn);
  k_pack_w   <<<4, 256, 0, stream>>>(Wf, Wp);
  k_gemm_wmma<<<N_GRAPHS / 32, 256, 0, stream>>>(pooledn, Wp, bf, (float*)d_out);
}
